// WaveNet_16423954940342
// MI455X (gfx1250) — matrix operand order repaired and checked by reading, not compiled, not run on hardware
//
#include <hip/hip_runtime.h>
#include <hip/hip_bf16.h>
#include <math.h>

typedef __attribute__((ext_vector_type(16))) __bf16        v16bf;
typedef __attribute__((ext_vector_type(8)))  float         v8f;
typedef __attribute__((ext_vector_type(4)))  unsigned int  v4u;

#define TLEN   16384
#define BATCH  8
#define RC     120     // residual channels
#define RCP    128     // padded residual channels
#define SC     240     // skip channels
#define NQ     256     // quant levels
#define TILE   64      // time samples per workgroup
#define HALO   128     // max dilation (zero-padded guard rows before t=0)
#define TROWS  (TLEN + HALO)   // rows per batch in x buffers
#define ZST    136     // zs LDS row stride (272B: 16B aligned, 4-dword bank step)
#define HST    264     // head LDS row stride (528B: 16B aligned)

__device__ __forceinline__ v8f wmma_bf16(v16bf a, v16bf b, v8f c) {
  return __builtin_amdgcn_wmma_f32_16x16x32_bf16(false, a, false, b, (short)0, c,
                                                 false, false);
}

union frag_u { v16bf v; v4u u[2]; };
union pk8_u  { v4u u; __bf16 h[8]; };

// 32 contiguous bytes -> bf16 fragment half-K slice (2x b128)
__device__ __forceinline__ v16bf load_frag_ptr(const __bf16* p) {
  const v4u* q = (const v4u*)p;
  frag_u t; t.u[0] = q[0]; t.u[1] = q[1];
  return t.v;
}

// Pre-packed A fragment: 1KB per fragment, lane-contiguous.
__device__ __forceinline__ v16bf load_a_packed(const __bf16* __restrict__ frags,
                                               int fidx, int lane) {
  return load_frag_ptr(frags + (size_t)fidx * 512 + lane * 16);
}

__device__ __forceinline__ float sigmoid_fast(float x) {
  return __builtin_amdgcn_rcpf(1.0f + __expf(-x));
}

// ---------------- weight pre-pack into WMMA A-fragment layout ----------------
// frag order: (((tile*nchunk)+c)*kw + tp) * 512 + lane*16 + j
__global__ void pack_a_frags(const float* __restrict__ W, __bf16* __restrict__ out,
                             int outc, int inc, int kw, int ntile, int nchunk) {
  int idx = blockIdx.x * blockDim.x + threadIdx.x;
  int total = ntile * nchunk * kw * 512;
  if (idx >= total) return;
  int j    = idx & 15;
  int lane = (idx >> 4) & 31;
  int f    = idx >> 9;
  int tp   = f % kw;          f /= kw;
  int c    = f % nchunk;      f /= nchunk;
  int m0   = f * 16;
  int hf = lane >> 4, lm = lane & 15;
  int v = j >> 1, p = j & 1;
  int k = hf * 16 + v * 2 + p;
  int mm = m0 + lm, kk = c * 32 + k;
  float val = 0.f;
  if (mm < outc && kk < inc) val = W[((size_t)mm * inc + kk) * kw + tp];
  out[idx] = (__bf16)val;
}

// ---------------- input conv + halo zeroing ----------------------------------
__global__ void wavenet_input(const float* __restrict__ wave,
                              const float* __restrict__ Win,
                              const float* __restrict__ bin,
                              __bf16* __restrict__ x) {
  int idx = blockIdx.x * blockDim.x + threadIdx.x;
  int c = idx & (RCP - 1);
  int t = idx >> 7;
  int b = blockIdx.y;
  float v = 0.f;
  if (c < RC) v = Win[c] * wave[(size_t)b * TLEN + t] + bin[c];
  x[((size_t)b * TROWS + HALO + t) * RCP + c] = (__bf16)v;
}

__global__ void wavenet_zero_halo(__bf16* __restrict__ xA, __bf16* __restrict__ xB) {
  int idx = blockIdx.x * blockDim.x + threadIdx.x;  // HALO*RCP per batch
  int b = blockIdx.y;
  size_t o = (size_t)b * TROWS * RCP + idx;
  xA[o] = (__bf16)0.f;
  xB[o] = (__bf16)0.f;
}

__global__ void wavenet_zero_skip(float* __restrict__ skip) {
  int t = blockIdx.x * blockDim.x + threadIdx.x;
  int c = blockIdx.y, b = blockIdx.z;
  skip[((size_t)b * SC + c) * TLEN + t] = 0.f;
}

// ---------------- one residual block ----------------------------------------
__global__ __launch_bounds__(256) void wavenet_block(
    const __bf16* __restrict__ x_in, __bf16* __restrict__ x_out,
    float* __restrict__ skip,
    const __bf16* __restrict__ fF, const float* __restrict__ bfv,
    const __bf16* __restrict__ fG, const float* __restrict__ bgv,
    const __bf16* __restrict__ fR, const float* __restrict__ brv,
    const __bf16* __restrict__ fS, const float* __restrict__ bsv,
    int dil) {
  __shared__ __attribute__((aligned(16))) __bf16 zs[TILE][ZST];

  const int b   = blockIdx.y;
  const int t0  = blockIdx.x * TILE;
  const int tid = threadIdx.x;
  const __bf16* xb = x_in + ((size_t)b * TROWS + HALO) * RCP;  // t=0 base

  if (t0 + TILE < TLEN)  // speculative prefetch of next time tile
    __builtin_prefetch(xb + (size_t)(t0 + TILE) * RCP + tid * 32, 0, 0);

  const int wv = tid >> 5, lane = tid & 31;
  const int hf = lane >> 4, lm = lane & 15;

  // ---- phase 1: f/g dilated convs + gated activation -> zs ----
  {
    const int m0 = wv * 16;
    v16bf af[4][2], ag[4][2];
#pragma unroll
    for (int c = 0; c < 4; ++c)
#pragma unroll
      for (int tp = 0; tp < 2; ++tp) {
        int fidx = ((wv * 4 + c) * 2 + tp);
        af[c][tp] = load_a_packed(fF, fidx, lane);
        ag[c][tp] = load_a_packed(fG, fidx, lane);
      }
#pragma unroll
    for (int nt = 0; nt < 4; ++nt) {
      v8f accf = {}, accg = {};
      // per-lane row pointers; halo rows make t-d unconditionally valid
      const __bf16* r1 = xb + (size_t)(t0 + nt * 16 + lm) * RCP + hf * 16;
      const __bf16* r0 = r1 - (size_t)dil * RCP;
#pragma unroll
      for (int c = 0; c < 4; ++c) {
        v16bf b0 = load_frag_ptr(r0 + c * 32);  // tap 0: x[t-d]
        v16bf b1 = load_frag_ptr(r1 + c * 32);  // tap 1: x[t]
        accf = wmma_bf16(af[c][0], b0, accf);
        accg = wmma_bf16(ag[c][0], b0, accg);
        accf = wmma_bf16(af[c][1], b1, accf);
        accg = wmma_bf16(ag[c][1], b1, accg);
      }
      pk8_u pk;
#pragma unroll
      for (int v = 0; v < 8; ++v) {
        int m = m0 + v + 8 * hf;               // C/D: M = vgpr + 8*(lane>=16)
        float bb = (m < RC) ? bfv[m] : 0.f;
        float gb = (m < RC) ? bgv[m] : 0.f;
        float f = accf[v] + bb;
        float g = accg[v] + gb;
        float th = 2.0f * sigmoid_fast(2.0f * f) - 1.0f;   // tanh
        float z  = th * sigmoid_fast(g);
        pk.h[v] = (__bf16)z;                   // padded rows: tanh(0)*0.5 = 0
      }
      *(v4u*)&zs[nt * 16 + lm][m0 + 8 * hf] = pk.u;   // ds_store_b128
    }
  }
  __syncthreads();

  // ---- phase 2: skip += Ws z + bs ; x_out = x_in + Wr z + br ----
  __bf16* xo = x_out + ((size_t)b * TROWS + HALO) * RCP;
  for (int job = wv; job < 23; job += 8) {
    const bool isSkip = job < 15;
    const int m0      = isSkip ? job * 16 : (job - 15) * 16;
    const __bf16* fW  = isSkip ? fS : fR;
    const float* bia  = isSkip ? bsv : brv;
    const int tilei   = isSkip ? job : (job - 15);
    v16bf aw[4];
#pragma unroll
    for (int c = 0; c < 4; ++c)
      aw[c] = load_a_packed(fW, tilei * 4 + c, lane);
#pragma unroll
    for (int nt = 0; nt < 4; ++nt) {
      v8f acc = {};
      const int trow = nt * 16 + lm;
#pragma unroll
      for (int c = 0; c < 4; ++c) {
        v16bf bz = load_frag_ptr(&zs[trow][c * 32 + hf * 16]);  // 2x ds_load_b128
        acc = wmma_bf16(aw[c], bz, acc);
      }
      const int t = t0 + trow;
      if (isSkip) {
#pragma unroll
        for (int v = 0; v < 8; ++v) {
          int m = m0 + v + 8 * hf;
          if (m < SC) {
            size_t o = ((size_t)b * SC + m) * TLEN + t;
            skip[o] += acc[v] + bia[m];
          }
        }
      } else {
        pk8_u xi, xov;
        xi.u = *(const v4u*)(xb + (size_t)t * RCP + m0 + 8 * hf);
#pragma unroll
        for (int v = 0; v < 8; ++v) {
          int m = m0 + v + 8 * hf;
          float xv = 0.f;
          if (m < RC) xv = (float)xi.h[v] + acc[v] + bia[m];
          xov.h[v] = (__bf16)xv;
        }
        *(v4u*)(xo + (size_t)t * RCP + m0 + 8 * hf) = xov.u;
      }
    }
  }
}

// ---------------- head: relu -> W1 -> relu -> W2 ----------------------------
__global__ __launch_bounds__(512) void wavenet_head(
    const float* __restrict__ skip,
    const __bf16* __restrict__ f1, const float* __restrict__ b1,
    const __bf16* __restrict__ f2, const float* __restrict__ b2,
    float* __restrict__ out) {
  __shared__ __attribute__((aligned(16))) __bf16 sbuf[TILE][HST];
  const int b = blockIdx.y, t0 = blockIdx.x * TILE;
  const int tid = threadIdx.x, wv = tid >> 5, lane = tid & 31;
  const int hf = lane >> 4, lm = lane & 15;
  const int m0 = wv * 16;

  // stage relu(skip) -> sbuf [t][256ch] bf16 (t-coalesced, zero-padded ch)
  for (int idx = tid; idx < TILE * NQ; idx += 512) {
    int t = idx & (TILE - 1);
    int k = idx >> 6;
    float v = 0.f;
    if (k < SC) v = fmaxf(skip[((size_t)b * SC + k) * TLEN + t0 + t], 0.f);
    sbuf[t][k] = (__bf16)v;
  }
  __syncthreads();

  // phase 1: h1 = relu(W1 @ relu(skip) + b1), kept in registers
  v8f hr[4];
  {
    v16bf a1[8];
#pragma unroll
    for (int c = 0; c < 8; ++c)
      a1[c] = load_a_packed(f1, wv * 8 + c, lane);
#pragma unroll
    for (int nt = 0; nt < 4; ++nt) {
      v8f acc = {};
      const int trow = nt * 16 + lm;
#pragma unroll
      for (int c = 0; c < 8; ++c) {
        v16bf bm = load_frag_ptr(&sbuf[trow][c * 32 + hf * 16]);
        acc = wmma_bf16(a1[c], bm, acc);
      }
      hr[nt] = acc;
    }
  }
  __syncthreads();

  // write h1 into sbuf (bias + relu + pack b128)
#pragma unroll
  for (int nt = 0; nt < 4; ++nt) {
    pk8_u pk;
#pragma unroll
    for (int v = 0; v < 8; ++v) {
      int m = m0 + v + 8 * hf;
      pk.h[v] = (__bf16)fmaxf(hr[nt][v] + b1[m], 0.f);
    }
    *(v4u*)&sbuf[nt * 16 + lm][m0 + 8 * hf] = pk.u;
  }
  __syncthreads();

  // phase 2: logits = W2 @ h1 + b2 -> d_out
  {
    v16bf a2[8];
#pragma unroll
    for (int c = 0; c < 8; ++c)
      a2[c] = load_a_packed(f2, wv * 8 + c, lane);
#pragma unroll
    for (int nt = 0; nt < 4; ++nt) {
      v8f acc = {};
      const int trow = nt * 16 + lm;
#pragma unroll
      for (int c = 0; c < 8; ++c) {
        v16bf bm = load_frag_ptr(&sbuf[trow][c * 32 + hf * 16]);
        acc = wmma_bf16(a2[c], bm, acc);
      }
      const int t = t0 + trow;
#pragma unroll
      for (int v = 0; v < 8; ++v) {
        int m = m0 + v + 8 * hf;
        out[((size_t)b * NQ + m) * TLEN + t] = acc[v] + b2[m];
      }
    }
  }
}

extern "C" void kernel_launch(void* const* d_in, const int* in_sizes, int n_in,
                              void* d_out, int out_size, void* d_ws, size_t ws_size,
                              hipStream_t stream) {
  (void)in_sizes; (void)n_in; (void)out_size; (void)ws_size;
  const float* wave = (const float*)d_in[0];
  const float* Win  = (const float*)d_in[1];
  const float* bin  = (const float*)d_in[2];
  const float* Wf   = (const float*)d_in[3];
  const float* bfv  = (const float*)d_in[4];
  const float* Wg   = (const float*)d_in[5];
  const float* bgv  = (const float*)d_in[6];
  const float* Wr   = (const float*)d_in[7];
  const float* brv  = (const float*)d_in[8];
  const float* Ws   = (const float*)d_in[9];
  const float* bsv  = (const float*)d_in[10];
  const float* W1   = (const float*)d_in[11];
  const float* b1   = (const float*)d_in[12];
  const float* W2   = (const float*)d_in[13];
  const float* b2   = (const float*)d_in[14];

  char* ws = (char*)d_ws;
  const size_t xbytes  = (size_t)BATCH * TROWS * RCP * 2;      // ~32.25 MB
  const size_t skbytes = (size_t)BATCH * SC * TLEN * 4;        // 120 MB
  const size_t fFsz = 8 * 4 * 2 * 512 * 2;                     // 64 KB
  const size_t fGsz = fFsz;
  const size_t fRsz = 8 * 4 * 1 * 512 * 2;                     // 32 KB
  const size_t fSsz = 15 * 4 * 1 * 512 * 2;                    // 60 KB
  const size_t blkF = fFsz + fGsz + fRsz + fSsz;               // 220 KB / block

  __bf16* xA   = (__bf16*)ws;
  __bf16* xB   = (__bf16*)(ws + xbytes);
  float*  skip = (float*)(ws + 2 * xbytes);
  char*   fb   = ws + 2 * xbytes + skbytes;                    // frag base
  __bf16* fW1  = (__bf16*)(fb + 16 * blkF);
  __bf16* fW2  = (__bf16*)(fb + 16 * blkF + (size_t)16 * 8 * 512 * 2);

  auto pack = [&](const float* W, __bf16* dst, int outc, int inc, int kw,
                  int ntile, int nchunk) {
    int total = ntile * nchunk * kw * 512;
    pack_a_frags<<<dim3((total + 255) / 256), 256, 0, stream>>>(
        W, dst, outc, inc, kw, ntile, nchunk);
  };
  for (int i = 0; i < 16; ++i) {
    char* base = fb + (size_t)i * blkF;
    pack(Wf + (size_t)i * RC * RC * 2, (__bf16*)base,                        RC, RC, 2, 8, 4);
    pack(Wg + (size_t)i * RC * RC * 2, (__bf16*)(base + fFsz),               RC, RC, 2, 8, 4);
    pack(Wr + (size_t)i * RC * RC,     (__bf16*)(base + fFsz + fGsz),        RC, RC, 1, 8, 4);
    pack(Ws + (size_t)i * SC * RC,     (__bf16*)(base + fFsz + fGsz + fRsz), SC, RC, 1, 15, 4);
  }
  pack(W1, fW1, NQ, SC, 1, 16, 8);
  pack(W2, fW2, NQ, NQ, 1, 16, 8);

  wavenet_input<<<dim3(TLEN * RCP / 256, BATCH), 256, 0, stream>>>(wave, Win, bin, xA);
  wavenet_zero_halo<<<dim3(HALO * RCP / 256, BATCH), 256, 0, stream>>>(xA, xB);
  wavenet_zero_skip<<<dim3(TLEN / 256, SC, BATCH), 256, 0, stream>>>(skip);

  for (int i = 0; i < 16; ++i) {
    int dil = 1 << (i % 8);
    const __bf16* xi = (i & 1) ? xB : xA;
    __bf16*       xo = (i & 1) ? xA : xB;
    char* base = fb + (size_t)i * blkF;
    wavenet_block<<<dim3(TLEN / TILE, BATCH), 256, 0, stream>>>(
        xi, xo, skip,
        (const __bf16*)base,                        bfv + (size_t)i * RC,
        (const __bf16*)(base + fFsz),               bgv + (size_t)i * RC,
        (const __bf16*)(base + fFsz + fGsz),        brv + (size_t)i * RC,
        (const __bf16*)(base + fFsz + fGsz + fRsz), bsv + (size_t)i * SC,
        dil);
  }

  wavenet_head<<<dim3(TLEN / TILE, BATCH), 512, 0, stream>>>(
      skip, fW1, b1, fW2, b2, (float*)d_out);
}
